// MultiHeadAttention_9620726743623
// MI455X (gfx1250) — matrix operand order repaired and checked by reading, not compiled, not run on hardware
//
#include <hip/hip_runtime.h>
#include <hip/hip_bf16.h>

// ---------------------------------------------------------------------------
// MHA forward for gfx1250 (CDNA5, wave32, WMMA bf16 16x16x32, f32 accum)
// d_out = [ out (B*S*D f32) | attn (B*H*S*S f32) ]
// ---------------------------------------------------------------------------

#define D_MODEL 1024
#define NUM_HEADS 16
#define DEPTH 64
#define BATCH 2
#define SEQ 2048
#define TOKENS (BATCH * SEQ)          // 4096

typedef __attribute__((ext_vector_type(16))) __bf16    v16bf;
typedef __attribute__((ext_vector_type(2)))  __bf16    v2bf;
typedef __attribute__((ext_vector_type(8)))  float     v8f;
typedef __attribute__((ext_vector_type(4)))  unsigned  uv4;
typedef __attribute__((ext_vector_type(4)))  float     fv4;

union ABu { v16bf v; unsigned u[8]; uv4 q[2]; };

// hardware f32->bf16 (RTNE) — lowered by clang to packed cvt ops
static __device__ __forceinline__ unsigned pk_bf16(float lo, float hi) {
  v2bf t = { (__bf16)lo, (__bf16)hi };
  return __builtin_bit_cast(unsigned, t);
}

static __device__ __forceinline__ unsigned short bf16_of(float f) {
  __bf16 h = (__bf16)f;
  return __builtin_bit_cast(unsigned short, h);
}

static __device__ __forceinline__ v8f bwmma(const ABu& a, const ABu& b, v8f c) {
  return __builtin_amdgcn_wmma_f32_16x16x32_bf16(false, a.v, false, b.v,
                                                 (short)0, c, false, false);
}

// ---------------------------------------------------------------------------
// Weight convert + transpose:  W[k][n] f32  ->  WT[n][k] bf16
// ---------------------------------------------------------------------------
__global__ __launch_bounds__(256) void k_wt(const float* __restrict__ W,
                                            unsigned short* __restrict__ WT) {
  int idx = blockIdx.x * 256 + threadIdx.x;          // 0 .. 1M-1
  int k = idx >> 10, n = idx & 1023;
  WT[(size_t)n * 1024 + k] = bf16_of(W[idx]);
}

// ---------------------------------------------------------------------------
// GEMM tile loaders (A: 16x32 bf16 layout; B: 32x16 from transposed weights)
// ---------------------------------------------------------------------------
static __device__ __forceinline__ void load_a(ABu& a, const void* Aptr,
                                              int a_bf16, int row, int k0,
                                              int half) {
  if (a_bf16) {
    const unsigned short* Ar =
        (const unsigned short*)Aptr + (size_t)row * 1024 + k0;
    a.q[0] = *(const uv4*)(Ar + 8 * half);        // K = 8h .. 8h+7
    a.q[1] = *(const uv4*)(Ar + 16 + 8 * half);   // K = 16+8h .. +7
  } else {
    const float* Af = (const float*)Aptr + (size_t)row * 1024 + k0;
    fv4 x0 = *(const fv4*)(Af + 8 * half);
    fv4 x1 = *(const fv4*)(Af + 8 * half + 4);
    fv4 y0 = *(const fv4*)(Af + 16 + 8 * half);
    fv4 y1 = *(const fv4*)(Af + 16 + 8 * half + 4);
    a.u[0] = pk_bf16(x0.x, x0.y); a.u[1] = pk_bf16(x0.z, x0.w);
    a.u[2] = pk_bf16(x1.x, x1.y); a.u[3] = pk_bf16(x1.z, x1.w);
    a.u[4] = pk_bf16(y0.x, y0.y); a.u[5] = pk_bf16(y0.z, y0.w);
    a.u[6] = pk_bf16(y1.x, y1.y); a.u[7] = pk_bf16(y1.z, y1.w);
  }
}

static __device__ __forceinline__ void load_b4(ABu (&b)[4],
                                               const unsigned short* WT,
                                               int n0, int l15, int k0,
                                               int half) {
#pragma unroll
  for (int nt = 0; nt < 4; ++nt) {
    const unsigned short* Br =
        WT + (size_t)(n0 + nt * 16 + l15) * 1024 + k0 + 8 * half;
    b[nt].q[0] = *(const uv4*)Br;
    b[nt].q[1] = *(const uv4*)(Br + 16);
  }
}

// ---------------------------------------------------------------------------
// Generic GEMM:  C[m][n] = A(MxK) @ WT^T + bias, M=4096, N=K=1024
//   a_bf16  : 0 = A is f32 row-major, 1 = A is bf16 row-major
//   out_mode: 0 = bf16 head-split [b][h][s][d]
//             1 = bf16 head-split transposed [b][h][d][s]   (for V)
//             2 = f32 row-major [t][n]                      (final out)
// Each wave owns a 16x64 output strip; K-loop of 32, double-buffered.
// ---------------------------------------------------------------------------
__global__ __launch_bounds__(256) void k_proj(const void* __restrict__ Aptr,
                                              const unsigned short* __restrict__ WT,
                                              const float* __restrict__ bias,
                                              void* __restrict__ Out,
                                              int a_bf16, int out_mode) {
  const int wave = blockIdx.x * (blockDim.x >> 5) + (threadIdx.x >> 5);
  const int lane = threadIdx.x & 31;
  const int half = lane >> 4, l15 = lane & 15;
  const int m0 = (wave >> 4) * 16;     // 256 m-tiles
  const int n0 = (wave & 15) * 64;     // 16 n-strips
  const int row = m0 + l15;

  v8f acc[4] = {};
  ABu aA, aB;
  ABu bA[4], bB[4];

  load_a(aA, Aptr, a_bf16, row, 0, half);
  load_b4(bA, WT, n0, l15, 0, half);

  for (int k0 = 0; k0 < 1024; k0 += 64) {
    // prefetch odd stage
    load_a(aB, Aptr, a_bf16, row, k0 + 32, half);
    load_b4(bB, WT, n0, l15, k0 + 32, half);
#pragma unroll
    for (int nt = 0; nt < 4; ++nt) acc[nt] = bwmma(aA, bA[nt], acc[nt]);
    // prefetch next even stage
    if (k0 + 64 < 1024) {
      load_a(aA, Aptr, a_bf16, row, k0 + 64, half);
      load_b4(bA, WT, n0, l15, k0 + 64, half);
    }
#pragma unroll
    for (int nt = 0; nt < 4; ++nt) acc[nt] = bwmma(aB, bB[nt], acc[nt]);
  }

#pragma unroll
  for (int nt = 0; nt < 4; ++nt) {
    const int n = n0 + nt * 16 + l15;
    const float bv = bias[n];
#pragma unroll
    for (int v = 0; v < 8; ++v) {
      const int m = m0 + v + 8 * half;
      const float val = acc[nt][v] + bv;
      if (out_mode == 2) {
        ((float*)Out)[(size_t)m * 1024 + n] = val;
      } else {
        const int b = m >> 11, s = m & 2047, h = n >> 6, d = n & 63;
        size_t addr;
        if (out_mode == 0)
          addr = (((size_t)(b * 16 + h) * 2048) + s) * 64 + d;  // [b][h][s][d]
        else
          addr = (((size_t)(b * 16 + h) * 64) + d) * 2048 + s;  // [b][h][d][s]
        ((unsigned short*)Out)[addr] = bf16_of(val);
      }
    }
  }
}

// ---------------------------------------------------------------------------
// Fused attention: logits -> causal mask -> softmax -> attn store -> attn@V
// grid = (S/16, B*H); block = 256 (8 waves). One block = 16 query rows.
// ---------------------------------------------------------------------------
static __device__ __forceinline__ void load_kb(ABu& b0, ABu& b1,
                                               const unsigned short* kh,
                                               size_t base, int c0, int l15,
                                               int half) {
  const unsigned short* Br = kh + base + (size_t)(c0 + l15) * 64;
  b0.q[0] = *(const uv4*)(Br + 8 * half);
  b0.q[1] = *(const uv4*)(Br + 16 + 8 * half);
  b1.q[0] = *(const uv4*)(Br + 32 + 8 * half);
  b1.q[1] = *(const uv4*)(Br + 48 + 8 * half);
}

__global__ __launch_bounds__(256) void k_attn(const unsigned short* __restrict__ qh,
                                              const unsigned short* __restrict__ kh,
                                              const unsigned short* __restrict__ vhT,
                                              float* __restrict__ attn_out,
                                              unsigned short* __restrict__ O) {
  __shared__ float lds[16 * 2048];       // 128 KB logits / attn rows
  __shared__ float red[16 * 16];
  __shared__ float rowm[16];
  __shared__ float rowinv[16];
  __shared__ float opart[2 * 16 * 64];   // split-K partials

  const int bh = blockIdx.y;             // b*16 + h
  const int q0 = blockIdx.x * 16;
  const int tid = (int)threadIdx.x;
  const int w = tid >> 5, lane = tid & 31, half = lane >> 4, l15 = lane & 15;
  const size_t base = (size_t)bh * (2048 * 64);

  // -------- Phase 1: logits (16 x 2048) into LDS --------
  ABu a0, a1;
  {
    const unsigned short* Ar = qh + base + (size_t)(q0 + l15) * 64;
    a0.q[0] = *(const uv4*)(Ar + 8 * half);
    a0.q[1] = *(const uv4*)(Ar + 16 + 8 * half);
    a1.q[0] = *(const uv4*)(Ar + 32 + 8 * half);
    a1.q[1] = *(const uv4*)(Ar + 48 + 8 * half);
  }
  const float scale = 0.125f;            // 1/sqrt(64)
  const int jt_last = q0 >> 4;           // last tile touching the diagonal

  {
    ABu c0b, c1b, n0b, n1b;
    int jt = w;
    if (jt <= jt_last) load_kb(c0b, c1b, kh, base, jt * 16, l15, half);
    for (; jt <= jt_last; jt += 8) {
      const int jn = jt + 8;
      if (jn <= jt_last) load_kb(n0b, n1b, kh, base, jn * 16, l15, half);
      v8f acc = {};
      acc = bwmma(a0, c0b, acc);
      acc = bwmma(a1, c1b, acc);
      const int c0 = jt * 16;
#pragma unroll
      for (int v = 0; v < 8; ++v) {
        const int m = v + 8 * half, n = l15;
        float val = acc[v] * scale;
        if (c0 + n > q0 + m) val -= 1e9f;  // causal mask (diagonal tile)
        lds[m * 2048 + c0 + n] = val;
      }
      c0b = n0b; c1b = n1b;
    }
  }
  // fully masked tiles: plain fill, no loads
  for (int jt = w; jt < 128; jt += 8) {
    if (jt <= jt_last) continue;
    const int c0 = jt * 16;
#pragma unroll
    for (int v = 0; v < 8; ++v)
      lds[(v + 8 * half) * 2048 + c0 + l15] = -1e9f;
  }
  __syncthreads();

  // -------- Phase 2: softmax over each of the 16 rows --------
  const int r = tid >> 4, l16 = tid & 15;
  float mx = -3.0e38f;
  for (int c = l16; c < 2048; c += 16) mx = fmaxf(mx, lds[r * 2048 + c]);
  red[r * 16 + l16] = mx;
  __syncthreads();
  if (tid < 16) {
    float m2 = red[tid * 16];
#pragma unroll
    for (int i = 1; i < 16; ++i) m2 = fmaxf(m2, red[tid * 16 + i]);
    rowm[tid] = m2;
  }
  __syncthreads();
  const float rm = rowm[r];
  float sum = 0.f;
  for (int c = l16; c < 2048; c += 16) {
    float e = __expf(lds[r * 2048 + c] - rm);
    lds[r * 2048 + c] = e;
    sum += e;
  }
  red[r * 16 + l16] = sum;
  __syncthreads();
  if (tid < 16) {
    float s2 = 0.f;
#pragma unroll
    for (int i = 0; i < 16; ++i) s2 += red[tid * 16 + i];
    rowinv[tid] = 1.0f / s2;
  }
  __syncthreads();

  // normalize in LDS + coalesced f32 attn store
  float* ab = attn_out + ((size_t)bh * 2048 + q0) * 2048;
  for (int idx = tid; idx < 16 * 2048; idx += 256) {
    const float v = lds[idx] * rowinv[idx >> 11];
    lds[idx] = v;
    ab[idx] = v;
  }
  __syncthreads();

  // -------- Phase 3: O(16x64) = attn(16x2048) @ vh(2048x64), split-K x2 ----
  const int nt = w & 3;                  // output 16-col tile
  const int khalf = w >> 2;              // K half [khalf*1024, +1024)
  const int kc0 = khalf * 1024;
  float* op = opart + khalf * 1024 + nt * 16;
  if (kc0 <= q0 + 15) {
    const int lim = (q0 + 16 + 31) & ~31;               // beyond this attn==0
    const int kend = (kc0 + 1024 < lim) ? kc0 + 1024 : lim;
    v8f oacc = {};
    for (int kk = kc0; kk < kend; kk += 32) {
      // A tile from LDS attn rows (ds_load_b128 x4 per lane)
      ABu a;
      const float* Lr = lds + l15 * 2048 + kk + 8 * half;
      fv4 x0 = *(const fv4*)(Lr);
      fv4 x1 = *(const fv4*)(Lr + 4);
      fv4 y0 = *(const fv4*)(Lr + 16);
      fv4 y1 = *(const fv4*)(Lr + 20);
      a.u[0] = pk_bf16(x0.x, x0.y); a.u[1] = pk_bf16(x0.z, x0.w);
      a.u[2] = pk_bf16(x1.x, x1.y); a.u[3] = pk_bf16(x1.z, x1.w);
      a.u[4] = pk_bf16(y0.x, y0.y); a.u[5] = pk_bf16(y0.z, y0.w);
      a.u[6] = pk_bf16(y1.x, y1.y); a.u[7] = pk_bf16(y1.z, y1.w);
      ABu b;
      const unsigned short* Vr =
          vhT + base + (size_t)(nt * 16 + l15) * 2048 + kk + 8 * half;
      b.q[0] = *(const uv4*)Vr;
      b.q[1] = *(const uv4*)(Vr + 16);
      oacc = bwmma(a, b, oacc);
    }
#pragma unroll
    for (int v = 0; v < 8; ++v) op[(v + 8 * half) * 64 + l15] = oacc[v];
  } else {
    for (int m = half; m < 16; m += 2) op[m * 64 + l15] = 0.f;
  }
  __syncthreads();

  // reduce split-K partials, store O as bf16 head-merged [t][1024]
  const int b_ = bh >> 4, h_ = bh & 15;
  for (int idx = tid; idx < 1024; idx += 256) {
    const float s = opart[idx] + opart[1024 + idx];
    const int rr = idx >> 6, d = idx & 63;
    const size_t t = (size_t)b_ * 2048 + q0 + rr;
    O[t * 1024 + h_ * 64 + d] = bf16_of(s);
  }
}

// ---------------------------------------------------------------------------
extern "C" void kernel_launch(void* const* d_in, const int* in_sizes, int n_in,
                              void* d_out, int out_size, void* d_ws, size_t ws_size,
                              hipStream_t stream) {
  (void)in_sizes; (void)n_in; (void)out_size; (void)ws_size;

  const float* q   = (const float*)d_in[0];
  const float* k   = (const float*)d_in[1];
  const float* v   = (const float*)d_in[2];
  // d_in[3] = mask (causal, reproduced by index test in-kernel)
  const float* Wq  = (const float*)d_in[4];
  const float* bq  = (const float*)d_in[5];
  const float* Wk  = (const float*)d_in[6];
  const float* bk  = (const float*)d_in[7];
  const float* Wv  = (const float*)d_in[8];
  const float* bv  = (const float*)d_in[9];
  const float* Wo  = (const float*)d_in[10];
  const float* bo  = (const float*)d_in[11];

  unsigned short* wsb = (unsigned short*)d_ws;
  unsigned short* WqT = wsb;                         // 1M bf16 each
  unsigned short* WkT = wsb + (1u << 20);
  unsigned short* WvT = wsb + 2u * (1u << 20);
  unsigned short* WoT = wsb + 3u * (1u << 20);
  unsigned short* qh  = wsb + 4u * (1u << 20);       // 4M bf16 each
  unsigned short* kh  = wsb + 8u * (1u << 20);
  unsigned short* vhT = wsb + 12u * (1u << 20);
  unsigned short* O   = wsb + 16u * (1u << 20);

  float* out_f   = (float*)d_out;
  float* attn_f  = out_f + (size_t)TOKENS * D_MODEL; // after `out`

  // 1) weights -> transposed bf16
  k_wt<<<4096, 256, 0, stream>>>(Wq, WqT);
  k_wt<<<4096, 256, 0, stream>>>(Wk, WkT);
  k_wt<<<4096, 256, 0, stream>>>(Wv, WvT);
  k_wt<<<4096, 256, 0, stream>>>(Wo, WoT);

  // 2) projections (4096 waves = 512 blocks x 8 waves)
  k_proj<<<512, 256, 0, stream>>>(q, WqT, bq, qh,  0, 0);
  k_proj<<<512, 256, 0, stream>>>(k, WkT, bk, kh,  0, 0);
  k_proj<<<512, 256, 0, stream>>>(v, WvT, bv, vhT, 0, 1);

  // 3) fused attention (writes attn f32 + O bf16)
  k_attn<<<dim3(SEQ / 16, BATCH * NUM_HEADS), 256, 0, stream>>>(
      qh, kh, vhT, attn_f, O);

  // 4) output projection -> d_out f32
  k_proj<<<512, 256, 0, stream>>>(O, WoT, bo, out_f, 1, 2);
}
